// MemoryGenerator_72490458021936
// MI455X (gfx1250) — matrix operand order repaired and checked by reading, not compiled, not run on hardware
//
#include <hip/hip_runtime.h>
#include <stdint.h>

// ---- problem constants (match reference setup_inputs) ----
#define B_   128
#define NZ_  128
#define H_   512
#define H2_  1024
#define H3_  1536
#define V_   8192
#define S_   64
#define NEG_ 0.2f

typedef __bf16 bf16;
typedef __attribute__((ext_vector_type(16))) __bf16 v16bf;
typedef __attribute__((ext_vector_type(8)))  __bf16 bf16x8;
typedef __attribute__((ext_vector_type(8)))  float  v8f;

__device__ __forceinline__ float leakyf(float x){ return x > 0.f ? x : NEG_ * x; }
__device__ __forceinline__ float sigm(float x){ return 1.f / (1.f + expf(-x)); }

// ============================================================
// threefry2x32 (jax PRNG), 20 rounds
// ============================================================
__device__ __forceinline__ uint32_t rotl32(uint32_t v, int n){ return (v << n) | (v >> (32 - n)); }

__device__ __forceinline__ void tf2x32(uint32_t k0, uint32_t k1, uint32_t c0, uint32_t c1,
                                       uint32_t& o0, uint32_t& o1)
{
  const uint32_t ks2 = k0 ^ k1 ^ 0x1BD11BDAu;
  uint32_t x0 = c0 + k0, x1 = c1 + k1;
  const int ra[4] = {13, 15, 26, 6};
  const int rb[4] = {17, 29, 16, 24};
#pragma unroll
  for (int i = 0; i < 4; ++i){ x0 += x1; x1 = rotl32(x1, ra[i]); x1 ^= x0; }
  x0 += k1;  x1 += ks2 + 1u;
#pragma unroll
  for (int i = 0; i < 4; ++i){ x0 += x1; x1 = rotl32(x1, rb[i]); x1 ^= x0; }
  x0 += ks2; x1 += k0 + 2u;
#pragma unroll
  for (int i = 0; i < 4; ++i){ x0 += x1; x1 = rotl32(x1, ra[i]); x1 ^= x0; }
  x0 += k0;  x1 += k1 + 3u;
#pragma unroll
  for (int i = 0; i < 4; ++i){ x0 += x1; x1 = rotl32(x1, rb[i]); x1 ^= x0; }
  x0 += k1;  x1 += ks2 + 4u;
#pragma unroll
  for (int i = 0; i < 4; ++i){ x0 += x1; x1 = rotl32(x1, ra[i]); x1 ^= x0; }
  x0 += ks2; x1 += k0 + 5u;
  o0 = x0; o1 = x1;
}

// ============================================================
// WMMA bf16 GEMM:  C[M x N] = A[M x K] * Bw[N x K]^T + bias[N]
//  - A panel (32 rows x K) async-copied to LDS once per block
//  - B fragments ping-pong double-buffered (K unrolled by 2 -> no reg moves)
//  - block = 8 waves (2 M x 4 N), wave tile 16x64; K % 64 == 0 required
// ============================================================
__device__ __forceinline__ v16bf load_fragA_lds(const bf16* __restrict__ base, int ld,
                                                int row0, int k0, int lane)
{
  // ISA 16-bit A layout: lanes 0-15 row M, K[0..7]+K[16..23]; lanes 16-31 K[8..15]+K[24..31]
  int l  = lane & 15;
  int hi = lane >> 4;
  const bf16* p = base + (size_t)(row0 + l) * ld + k0 + hi * 8;
  union { v16bf v; bf16x8 h[2]; } u;
  u.h[0] = *(const bf16x8*)(p);
  u.h[1] = *(const bf16x8*)(p + 16);
  return u.v;
}

__device__ __forceinline__ v16bf load_fragB(const bf16* __restrict__ base, int ld,
                                            int n0, int k0, int lane)
{
  // ISA 16-bit B layout: lanes 0-15 col N, K[0..7]+K[16..23]; lanes 16-31 K[8..15]+K[24..31]
  int l  = lane & 15;
  int hi = lane >> 4;
  const bf16* p = base + (size_t)(n0 + l) * ld + k0 + hi * 8;
  union { v16bf v; bf16x8 h[2]; } u;
  u.h[0] = *(const bf16x8*)(p);
  u.h[1] = *(const bf16x8*)(p + 16);
  return u.v;
}

__global__ void __launch_bounds__(256) k_wmma_gemm(
    const bf16* __restrict__ A, const bf16* __restrict__ Bw,
    const float* __restrict__ bias, float* __restrict__ C,
    int M, int N, int K)
{
  extern __shared__ bf16 Atile[];            // 32 x K panel, row-major
  const int tid  = threadIdx.x;
  const int lane = tid & 31;
  const int wave = tid >> 5;
  const int wm = wave >> 2, wn = wave & 3;
  const int mBlock = blockIdx.x * 32;
  const int n0 = blockIdx.y * 256 + wn * 64;
  const int ar0 = wm * 16;                   // this wave's A rows within LDS panel

  // ---- stage A panel into LDS with CDNA5 async copies (ASYNCcnt) ----
  {
    const int chunksPerRow = K >> 3;          // 16-byte chunks per row
    const int totalChunks  = 32 * chunksPerRow;
    const uint32_t lbase = (uint32_t)(uintptr_t)Atile;   // LDS addr = low 32 bits (aperture rule)
    for (int c = tid; c < totalChunks; c += 256) {
      int row = c / chunksPerRow;
      int kc  = (c - row * chunksPerRow) << 3;
      uint64_t gp = (uint64_t)(uintptr_t)(A + (size_t)(mBlock + row) * K + kc);
      uint32_t la = lbase + (uint32_t)(row * K + kc) * 2u;
      asm volatile("global_load_async_to_lds_b128 %0, %1, off"
                   :: "v"(la), "v"(gp) : "memory");
    }
    asm volatile("s_wait_asynccnt 0x0" ::: "memory");
    __syncthreads();
  }

  v8f acc0 = {}, acc1 = {}, acc2 = {}, acc3 = {};

  // ping buffer: tile k0 = 0
  v16bf bA0 = load_fragB(Bw, K, n0 +  0, 0, lane);
  v16bf bA1 = load_fragB(Bw, K, n0 + 16, 0, lane);
  v16bf bA2 = load_fragB(Bw, K, n0 + 32, 0, lane);
  v16bf bA3 = load_fragB(Bw, K, n0 + 48, 0, lane);

  // steady state: two K-tiles per iteration, ping-pong (no register rotation)
  int k0 = 0;
  for (; k0 + 64 < K; k0 += 64) {
    // pong buffer: tile k0+32 (issued before WMMAs on ping)
    v16bf bB0 = load_fragB(Bw, K, n0 +  0, k0 + 32, lane);
    v16bf bB1 = load_fragB(Bw, K, n0 + 16, k0 + 32, lane);
    v16bf bB2 = load_fragB(Bw, K, n0 + 32, k0 + 32, lane);
    v16bf bB3 = load_fragB(Bw, K, n0 + 48, k0 + 32, lane);
    __builtin_prefetch(Bw + (size_t)(n0 + lane) * K + k0 + 96, 0, 1);

    v16bf a = load_fragA_lds(Atile, K, ar0, k0, lane);
    acc0 = __builtin_amdgcn_wmma_f32_16x16x32_bf16(false, a, false, bA0, (short)0, acc0, false, false);
    acc1 = __builtin_amdgcn_wmma_f32_16x16x32_bf16(false, a, false, bA1, (short)0, acc1, false, false);
    acc2 = __builtin_amdgcn_wmma_f32_16x16x32_bf16(false, a, false, bA2, (short)0, acc2, false, false);
    acc3 = __builtin_amdgcn_wmma_f32_16x16x32_bf16(false, a, false, bA3, (short)0, acc3, false, false);

    // refill ping with tile k0+64 (issued before WMMAs on pong)
    bA0 = load_fragB(Bw, K, n0 +  0, k0 + 64, lane);
    bA1 = load_fragB(Bw, K, n0 + 16, k0 + 64, lane);
    bA2 = load_fragB(Bw, K, n0 + 32, k0 + 64, lane);
    bA3 = load_fragB(Bw, K, n0 + 48, k0 + 64, lane);

    v16bf a2 = load_fragA_lds(Atile, K, ar0, k0 + 32, lane);
    acc0 = __builtin_amdgcn_wmma_f32_16x16x32_bf16(false, a2, false, bB0, (short)0, acc0, false, false);
    acc1 = __builtin_amdgcn_wmma_f32_16x16x32_bf16(false, a2, false, bB1, (short)0, acc1, false, false);
    acc2 = __builtin_amdgcn_wmma_f32_16x16x32_bf16(false, a2, false, bB2, (short)0, acc2, false, false);
    acc3 = __builtin_amdgcn_wmma_f32_16x16x32_bf16(false, a2, false, bB3, (short)0, acc3, false, false);
  }

  // epilogue: tiles k0 and k0+32 (k0 == K-64)
  {
    v16bf bB0 = load_fragB(Bw, K, n0 +  0, k0 + 32, lane);
    v16bf bB1 = load_fragB(Bw, K, n0 + 16, k0 + 32, lane);
    v16bf bB2 = load_fragB(Bw, K, n0 + 32, k0 + 32, lane);
    v16bf bB3 = load_fragB(Bw, K, n0 + 48, k0 + 32, lane);

    v16bf a = load_fragA_lds(Atile, K, ar0, k0, lane);
    acc0 = __builtin_amdgcn_wmma_f32_16x16x32_bf16(false, a, false, bA0, (short)0, acc0, false, false);
    acc1 = __builtin_amdgcn_wmma_f32_16x16x32_bf16(false, a, false, bA1, (short)0, acc1, false, false);
    acc2 = __builtin_amdgcn_wmma_f32_16x16x32_bf16(false, a, false, bA2, (short)0, acc2, false, false);
    acc3 = __builtin_amdgcn_wmma_f32_16x16x32_bf16(false, a, false, bA3, (short)0, acc3, false, false);

    v16bf a2 = load_fragA_lds(Atile, K, ar0, k0 + 32, lane);
    acc0 = __builtin_amdgcn_wmma_f32_16x16x32_bf16(false, a2, false, bB0, (short)0, acc0, false, false);
    acc1 = __builtin_amdgcn_wmma_f32_16x16x32_bf16(false, a2, false, bB1, (short)0, acc1, false, false);
    acc2 = __builtin_amdgcn_wmma_f32_16x16x32_bf16(false, a2, false, bB2, (short)0, acc2, false, false);
    acc3 = __builtin_amdgcn_wmma_f32_16x16x32_bf16(false, a2, false, bB3, (short)0, acc3, false, false);
  }

  // C/D layout: VGPR r -> M = m0 + r + 8*(lane>=16); N = n0 + (lane&15)
  const int l  = lane & 15;
  const int hi = lane >> 4;
  const int m0 = mBlock + wm * 16;
  v8f accs[4] = {acc0, acc1, acc2, acc3};
#pragma unroll
  for (int t = 0; t < 4; ++t) {
    int n = n0 + t * 16 + l;
    float bv = bias ? bias[n] : 0.f;
#pragma unroll
    for (int r = 0; r < 8; ++r) {
      int m = m0 + r + hi * 8;
      C[(size_t)m * N + n] = accs[t][r] + bv;
    }
  }
}

// ============================================================
// small helper kernels
// ============================================================
__global__ void __launch_bounds__(256) k_cvt_bf16(const float* __restrict__ src,
                                                  bf16* __restrict__ dst, int n)
{
  int t = blockIdx.x * 256 + threadIdx.x;
  if (t < n) dst[t] = (bf16)src[t];
}

// t = leaky(z @ z2h_w^T + z2h_b)   (128x128 @ 128x512)
__global__ void __launch_bounds__(256) k_zlin(const float* __restrict__ z,
                                              const float* __restrict__ w,
                                              const float* __restrict__ bias,
                                              float* __restrict__ out)
{
  int t = blockIdx.x * 256 + threadIdx.x;   // 0..65535
  int b = t >> 9, n = t & 511;
  const float* zr = z + b * NZ_;
  const float* wr = w + n * NZ_;
  float acc = bias[n];
#pragma unroll 4
  for (int k = 0; k < NZ_; ++k) acc += zr[k] * wr[k];
  out[t] = leakyf(acc);
}

// batchnorm over batch (rows), one block (128 threads) per column
__global__ void __launch_bounds__(128) k_bn(const float* __restrict__ x, int ldx,
    const float* __restrict__ g, const float* __restrict__ bta, int goff,
    float* __restrict__ outf, int ldof,
    bf16* __restrict__ outb, int ldob, int ocol)
{
  __shared__ float s1[128], s2[128];
  int c = blockIdx.x, b = threadIdx.x;
  float v = x[(size_t)b * ldx + c];
  s1[b] = v; s2[b] = v * v;
  __syncthreads();
  for (int s = 64; s > 0; s >>= 1) {
    if (b < s) { s1[b] += s1[b + s]; s2[b] += s2[b + s]; }
    __syncthreads();
  }
  float mu  = s1[0] * (1.f / 128.f);
  float var = s2[0] * (1.f / 128.f) - mu * mu;
  float sc  = g[goff + c] * rsqrtf(var + 1e-5f);
  float y   = sc * (v - mu) + bta[goff + c];
  if (outf) outf[(size_t)b * ldof + c] = y;
  if (outb) outb[(size_t)b * ldob + ocol + c] = (bf16)y;
}

// fused: x = leaky(emb[prev[b], c]); batchnorm with bn2[0:H]; write si cols [0,H)
__global__ void __launch_bounds__(128) k_embed_bn(const float* __restrict__ emb,
    const int* __restrict__ prev,
    const float* __restrict__ g, const float* __restrict__ bta,
    bf16* __restrict__ si)
{
  __shared__ float s1[128], s2[128];
  int c = blockIdx.x, b = threadIdx.x;
  float v = leakyf(emb[(size_t)prev[b] * H_ + c]);
  s1[b] = v; s2[b] = v * v;
  __syncthreads();
  for (int s = 64; s > 0; s >>= 1) {
    if (b < s) { s1[b] += s1[b + s]; s2[b] += s2[b + s]; }
    __syncthreads();
  }
  float mu  = s1[0] * (1.f / 128.f);
  float var = s2[0] * (1.f / 128.f) - mu * mu;
  float y   = g[c] * rsqrtf(var + 1e-5f) * (v - mu) + bta[c];
  si[(size_t)b * H2_ + c] = (bf16)y;
}

// gi_m = mem_i @ gm_wih^T + gm_bih  (rows of sm identical -> GEMV, 1536 outs)
__global__ void __launch_bounds__(256) k_gemv(const float* __restrict__ mem,
    const float* __restrict__ w, const float* __restrict__ bias,
    float* __restrict__ out)
{
  int n = blockIdx.x * 256 + threadIdx.x;   // 0..1535
  const float* wr = w + (size_t)n * H2_;
  float acc = bias[n];
#pragma unroll 4
  for (int k = 0; k < H2_; ++k) acc += mem[k] * wr[k];
  out[n] = acc;
}

// GRU pointwise (both GRUs) + SIM=4 cosine gate + leaky; updates carries
__global__ void __launch_bounds__(256) k_gru(const float* __restrict__ gi_x,
    const float* __restrict__ gh_x, const float* __restrict__ gim,
    const float* __restrict__ gh_m,
    float* __restrict__ hx, float* __restrict__ hm,
    bf16* __restrict__ hxb, bf16* __restrict__ hmb,
    float* __restrict__ outl)
{
  int t = blockIdx.x * 256 + threadIdx.x;   // 0..65535
  int b = t >> 9;
  int h = t & 511;
  size_t i3 = (size_t)b * H3_;

  float rx = sigm(gi_x[i3 + h]        + gh_x[i3 + h]);
  float zx = sigm(gi_x[i3 + 512 + h]  + gh_x[i3 + 512 + h]);
  float nx = tanhf(gi_x[i3 + 1024 + h] + rx * gh_x[i3 + 1024 + h]);
  float hx2 = (1.f - zx) * nx + zx * hx[t];

  float rm = sigm(gim[h]        + gh_m[i3 + h]);
  float zm = sigm(gim[512 + h]  + gh_m[i3 + 512 + h]);
  float nm = tanhf(gim[1024 + h] + rm * gh_m[i3 + 1024 + h]);
  float hm2 = (1.f - zm) * nm + zm * hm[t];

  // cosine similarity over groups of 4 consecutive h (stay within a wave)
  float dot = hx2 * hm2, na = hx2 * hx2, nb = hm2 * hm2;
  dot += __shfl_xor(dot, 1, 4); dot += __shfl_xor(dot, 2, 4);
  na  += __shfl_xor(na,  1, 4); na  += __shfl_xor(na,  2, 4);
  nb  += __shfl_xor(nb,  1, 4); nb  += __shfl_xor(nb,  2, 4);
  float gate = dot / fmaxf(sqrtf(na) * sqrtf(nb), 1e-8f);

  float o = gate * hx2 + (1.f - gate) * hm2;
  hx[t] = hx2; hm[t] = hm2;
  hxb[t] = (bf16)hx2; hmb[t] = (bf16)hm2;
  outl[t] = leakyf(o);
}

// gumbel-softmax + write probs (B,S,V layout) + argmax -> prev
__global__ void __launch_bounds__(256) k_sample(const float* __restrict__ logits,
    const float* __restrict__ temp, float* __restrict__ out,
    int* __restrict__ prev, int step)
{
  __shared__ float sred[256];
  __shared__ int   sidx[256];
  const int b = blockIdx.x, t = threadIdx.x;
  const float T = *temp;

  uint32_t fk0, fk1;                       // fold_in(key(42), step)
  tf2x32(0u, 42u, 0u, (uint32_t)step, fk0, fk1);

  const int nhalf = (B_ * V_) / 2;
  float y[32];
  float lmax = -3.4e38f; int lidx = 0x7fffffff;
#pragma unroll
  for (int j = 0; j < 32; ++j) {
    int v = j * 256 + t;
    int f = b * V_ + v;
    uint32_t o0, o1, bits;
    if (f < nhalf) { tf2x32(fk0, fk1, (uint32_t)f, (uint32_t)(f + nhalf), o0, o1); bits = o0; }
    else           { tf2x32(fk0, fk1, (uint32_t)(f - nhalf), (uint32_t)f, o0, o1); bits = o1; }
    float u01 = __uint_as_float((bits >> 9) | 0x3f800000u) - 1.f;
    float u = 1e-6f + u01 * (1.f - 2e-6f);
    u = fmaxf(u, 1e-6f);
    float gum = -logf(-logf(u));
    float yy = (logits[(size_t)b * V_ + v] + gum) / T;
    y[j] = yy;
    if (yy > lmax) { lmax = yy; lidx = v; }
  }
  sred[t] = lmax; sidx[t] = lidx; __syncthreads();
  for (int s = 128; s > 0; s >>= 1) {
    if (t < s) {
      float a = sred[t], c = sred[t + s];
      if (c > a || (c == a && sidx[t + s] < sidx[t])) { sred[t] = c; sidx[t] = sidx[t + s]; }
    }
    __syncthreads();
  }
  const float mx = sred[0];
  const int amax = sidx[0];
  __syncthreads();

  float lsum = 0.f;
#pragma unroll
  for (int j = 0; j < 32; ++j) { y[j] = expf(y[j] - mx); lsum += y[j]; }
  sred[t] = lsum; __syncthreads();
  for (int s = 128; s > 0; s >>= 1) { if (t < s) sred[t] += sred[t + s]; __syncthreads(); }
  const float inv = 1.f / sred[0];

  float* op = out + ((size_t)b * S_ + step) * V_;
#pragma unroll
  for (int j = 0; j < 32; ++j) op[j * 256 + t] = y[j] * inv;
  if (t == 0) prev[b] = amax;
}

// init carries + prev tokens
__global__ void __launch_bounds__(256) k_init(const float* __restrict__ z0,
    const bf16* __restrict__ z0b,
    float* __restrict__ hx, float* __restrict__ hm,
    bf16* __restrict__ hxb, bf16* __restrict__ hmb, int* __restrict__ prev)
{
  int t = blockIdx.x * 256 + threadIdx.x;
  if (t < B_ * H_) { hx[t] = z0[t]; hm[t] = z0[t]; hxb[t] = z0b[t]; hmb[t] = z0b[t]; }
  if (t < B_) prev[t] = V_ - 1;   // SOS
}

// ============================================================
// launcher
// ============================================================
extern "C" void kernel_launch(void* const* d_in, const int* in_sizes, int n_in,
                              void* d_out, int out_size, void* d_ws, size_t ws_size,
                              hipStream_t stream)
{
  const float* z      = (const float*)d_in[0];
  const float* temp   = (const float*)d_in[1];
  const float* z2h_w  = (const float*)d_in[2];
  const float* z2h_b  = (const float*)d_in[3];
  const float* bn1_g  = (const float*)d_in[4];
  const float* bn1_b  = (const float*)d_in[5];
  const float* emb    = (const float*)d_in[6];
  const float* bn2_g  = (const float*)d_in[7];
  const float* bn2_b  = (const float*)d_in[8];
  const float* memory = (const float*)d_in[9];
  const float* gx_wih = (const float*)d_in[10];
  const float* gx_whh = (const float*)d_in[11];
  const float* gx_bih = (const float*)d_in[12];
  const float* gx_bhh = (const float*)d_in[13];
  const float* gm_wih = (const float*)d_in[14];
  const float* gm_whh = (const float*)d_in[15];
  const float* gm_bih = (const float*)d_in[16];
  const float* gm_bhh = (const float*)d_in[17];
  const float* bn3_g  = (const float*)d_in[18];
  const float* bn3_b  = (const float*)d_in[19];
  const float* h2o_w  = (const float*)d_in[20];
  const float* h2o_b  = (const float*)d_in[21];
  (void)n_in; (void)in_sizes; (void)out_size; (void)ws_size;
  float* out = (float*)d_out;

  // ---- workspace carving (256B aligned) ----
  char* p = (char*)d_ws;
  auto carve = [&](size_t bytes) -> void* {
    void* r = (void*)p; p += (bytes + 255) & ~(size_t)255; return r;
  };
  float* tmp   = (float*)carve((size_t)B_ * H_ * 4);       // zlin / out_leaky (reused)
  float* z0    = (float*)carve((size_t)B_ * H_ * 4);
  float* hx    = (float*)carve((size_t)B_ * H_ * 4);
  float* hm    = (float*)carve((size_t)B_ * H_ * 4);
  float* gi_x  = (float*)carve((size_t)B_ * H3_ * 4);
  float* gh_x  = (float*)carve((size_t)B_ * H3_ * 4);
  float* gh_m  = (float*)carve((size_t)B_ * H3_ * 4);
  float* gim   = (float*)carve((size_t)H3_ * 4);
  float* logits= (float*)carve((size_t)B_ * V_ * 4);
  int*   prev  = (int*)  carve((size_t)B_ * 4);
  bf16* z0b    = (bf16*)carve((size_t)B_ * H_ * 2);
  bf16* si     = (bf16*)carve((size_t)B_ * H2_ * 2);       // [x_emb_bn | z0_bn] bf16
  bf16* hxb    = (bf16*)carve((size_t)B_ * H_ * 2);
  bf16* hmb    = (bf16*)carve((size_t)B_ * H_ * 2);
  bf16* outb   = (bf16*)carve((size_t)B_ * H_ * 2);
  bf16* w_gxih = (bf16*)carve((size_t)H3_ * H2_ * 2);
  bf16* w_gxhh = (bf16*)carve((size_t)H3_ * H_ * 2);
  bf16* w_gmhh = (bf16*)carve((size_t)H3_ * H_ * 2);
  bf16* w_h2o  = (bf16*)carve((size_t)V_ * H_ * 2);

  // ---- one-time (per launch) weight conversion to bf16 ----
  {
    int n;
    n = H3_ * H2_; k_cvt_bf16<<<(n + 255) / 256, 256, 0, stream>>>(gx_wih, w_gxih, n);
    n = H3_ * H_;  k_cvt_bf16<<<(n + 255) / 256, 256, 0, stream>>>(gx_whh, w_gxhh, n);
    n = H3_ * H_;  k_cvt_bf16<<<(n + 255) / 256, 256, 0, stream>>>(gm_whh, w_gmhh, n);
    n = V_ * H_;   k_cvt_bf16<<<(n + 255) / 256, 256, 0, stream>>>(h2o_w,  w_h2o,  n);
  }

  // ---- z0 = bn1(leaky(z @ z2h_w^T + b)); precompute bn2 z-half of si ----
  k_zlin<<<(B_ * H_) / 256, 256, 0, stream>>>(z, z2h_w, z2h_b, tmp);
  k_bn<<<H_, 128, 0, stream>>>(tmp, H_, bn1_g, bn1_b, 0, z0, H_, z0b, H_, 0);
  k_bn<<<H_, 128, 0, stream>>>(z0, H_, bn2_g, bn2_b, H_, nullptr, 0, si, H2_, H_);
  k_init<<<(B_ * H_) / 256, 256, 0, stream>>>(z0, z0b, hx, hm, hxb, hmb, prev);

  const size_t ldsA_K1024 = 32u * H2_ * sizeof(bf16);   // 64 KB
  const size_t ldsA_K512  = 32u * H_  * sizeof(bf16);   // 32 KB

  // ---- 64 sequential decode steps ----
  for (int i = 0; i < S_; ++i) {
    const float* mem_i = memory + (size_t)i * H2_;   // memory[0, i, :]

    k_embed_bn<<<H_, 128, 0, stream>>>(emb, prev, bn2_g, bn2_b, si);
    k_gemv<<<H3_ / 256, 256, 0, stream>>>(mem_i, gm_wih, gm_bih, gim);

    // gi_x = si @ gx_wih^T + bih       (M=128, N=1536, K=1024)
    k_wmma_gemm<<<dim3(B_ / 32, H3_ / 256), 256, ldsA_K1024, stream>>>(si, w_gxih, gx_bih, gi_x, B_, H3_, H2_);
    // gh_x = hx @ gx_whh^T + bhh       (M=128, N=1536, K=512)
    k_wmma_gemm<<<dim3(B_ / 32, H3_ / 256), 256, ldsA_K512, stream>>>(hxb, w_gxhh, gx_bhh, gh_x, B_, H3_, H_);
    // gh_m = hm @ gm_whh^T + bhh       (M=128, N=1536, K=512)
    k_wmma_gemm<<<dim3(B_ / 32, H3_ / 256), 256, ldsA_K512, stream>>>(hmb, w_gmhh, gm_bhh, gh_m, B_, H3_, H_);

    k_gru<<<(B_ * H_) / 256, 256, 0, stream>>>(gi_x, gh_x, gim, gh_m, hx, hm, hxb, hmb, tmp);
    k_bn<<<H_, 128, 0, stream>>>(tmp, H_, bn3_g, bn3_b, 0, nullptr, 0, outb, H_, 0);

    // logits = out @ h2o_w^T + h2o_b   (M=128, N=8192, K=512)
    k_wmma_gemm<<<dim3(B_ / 32, V_ / 256), 256, ldsA_K512, stream>>>(outb, w_h2o, h2o_b, logits, B_, V_, H_);

    k_sample<<<B_, 256, 0, stream>>>(logits, temp, out, prev, i);
  }
}
